// AutoCorrelation_58944131170420
// MI455X (gfx1250) — matrix operand order repaired and checked by reading, not compiled, not run on hardware
//
#include <hip/hip_runtime.h>
#include <hip/hip_bf16.h>

typedef __attribute__((ext_vector_type(16))) _Float16     v16h;
typedef __attribute__((ext_vector_type(8)))  _Float16     v8h;
typedef __attribute__((ext_vector_type(4)))  _Float16     v4h;
typedef __attribute__((ext_vector_type(8)))  float        v8f;
typedef __attribute__((ext_vector_type(4)))  unsigned int v4u;
typedef __attribute__((ext_vector_type(8)))  int          v8i;
typedef __attribute__((ext_vector_type(4)))  int          v4i;

constexpr int kB    = 16;
constexpr int kL    = 1536;
constexpr int kD    = 512;    // H*E = 8*64
constexpr int kNT   = 96;     // L / 16 time tiles
constexpr int kTOPK = 7;      // int(log(1536)) = 7
constexpr int kGR   = 8;      // gather tile rows (8 x 512 f32 = 16 KB)

#if __has_builtin(__builtin_amdgcn_tensor_load_to_lds) && \
    __has_builtin(__builtin_amdgcn_s_wait_tensorcnt)
#define AC_HAS_TDM 1
#else
#define AC_HAS_TDM 0
#endif

// ---------------------------------------------------------------------------
// Kernel 0: zero the mean_corr accumulator region of the workspace
// ---------------------------------------------------------------------------
__global__ void ac_zero_kernel(float* __restrict__ p, int n) {
  int i = blockIdx.x * blockDim.x + threadIdx.x;
  if (i < n) p[i] = 0.0f;
}

// ---------------------------------------------------------------------------
// Kernel 1: mean_corr[b, l] = (1/D) * sum_{t,d} Q[b,t,d] * K[b,(t-l)%L,d]
// One workgroup per (dt, b); dt = tile-diagonal offset (ti - tj) mod 96.
// 8 waves split the D=512 contraction (64 channels each -> 2 WMMA k-steps),
// each wave accumulates a 16x16 f32 tile over all 96 tj tiles, then waves +
// diagonals are reduced via LDS float atomics into 31 lag bins.
// ---------------------------------------------------------------------------
__global__ __launch_bounds__(256) void ac_corr_kernel(
    const float* __restrict__ q, const float* __restrict__ k,
    float* __restrict__ mean_corr) {
  const int dt   = blockIdx.x;        // 0..95
  const int b    = blockIdx.y;        // 0..15
  const int tid  = threadIdx.x;
  const int lane = tid & 31;
  const int wave = tid >> 5;          // 0..7 -> channel chunk wave*64

  __shared__ _Float16 Qs[16 * kD];    // 16 KB
  __shared__ _Float16 Ks[16 * kD];    // 16 KB
  __shared__ float    lagAcc[31];     // lag bins (i - j) in [-15, 15]

  if (tid < 31) lagAcc[tid] = 0.0f;

  const float* qb = q + (size_t)b * kL * kD;
  const float* kb = k + (size_t)b * kL * kD;

  v8f acc = {};

  // WMMA fragment layout used here (wave32; contiguous k, same for A and B):
  //  A 16x32 f16: lane holds row m = lane&15; halves[0..15] = K kbB..kbB+15,
  //               kbB = lane<16 ? 0 : 16 (loaded as two v8h).
  //  B 32x16 f16: lane holds col n = lane&15; halves[0..15] = K kbB..kbB+15,
  //               kbB = lane<16 ? 0 : 16.  (B[k][n] = Ktile[n][k])
  const int m     = lane & 15;
  const int kb0   = (lane < 16) ? 0 : 8;
  const int kbB   = (lane < 16) ? 0 : 16;
  const int dbase = wave * 64;

  for (int tj = 0; tj < kNT; ++tj) {
    int ti = tj + dt;
    if (ti >= kNT) ti -= kNT;
    const float4* qt4 = (const float4*)(qb + (size_t)(ti * 16) * kD);
    const float4* kt4 = (const float4*)(kb + (size_t)(tj * 16) * kD);

    // prefetch next iteration's tiles (global_prefetch_b8)
    if (tj + 1 < kNT) {
      int tin = tj + 1 + dt;
      if (tin >= kNT) tin -= kNT;
      __builtin_prefetch(qb + (size_t)(tin * 16) * kD + tid * 32, 0, 3);
      __builtin_prefetch(kb + (size_t)((tj + 1) * 16) * kD + tid * 32, 0, 3);
    }

    __syncthreads();  // previous iteration's fragment reads done
    // cooperative load + convert, vectorized: 2048 float4 per tile
#pragma unroll
    for (int r = 0; r < 8; ++r) {
      int    e4 = tid + 256 * r;
      float4 aq = qt4[e4];
      float4 ak = kt4[e4];
      v4h hq = {(_Float16)aq.x, (_Float16)aq.y, (_Float16)aq.z, (_Float16)aq.w};
      v4h hk = {(_Float16)ak.x, (_Float16)ak.y, (_Float16)ak.z, (_Float16)ak.w};
      *(v4h*)&Qs[e4 * 4] = hq;
      *(v4h*)&Ks[e4 * 4] = hk;
    }
    __syncthreads();

#pragma unroll
    for (int s = 0; s < 2; ++s) {
      const int       dk   = dbase + s * 32;
      const _Float16* qrow = &Qs[m * kD + dk];
      v8h  alo = *(const v8h*)(qrow + kbB);
      v8h  ahi = *(const v8h*)(qrow + kbB + 8);
      v16h a   = __builtin_shufflevector(alo, ahi, 0, 1, 2, 3, 4, 5, 6, 7, 8,
                                         9, 10, 11, 12, 13, 14, 15);
      v16h bf  = *(const v16h*)(&Ks[m * kD + dk + kbB]);
      acc = __builtin_amdgcn_wmma_f32_16x16x32_f16(
          /*neg_a=*/false, a, /*neg_b=*/false, bf,
          /*c_mod=*/(short)0, acc, /*reuse_a=*/false, /*reuse_b=*/false);
    }
  }
  __syncthreads();

  // C layout: VGPR r, lane: i = r + (lane<16?0:8), j = lane&15.
  const int   j     = lane & 15;
  const int   ibase = (lane < 16) ? 0 : 8;
  const float invD  = 1.0f / (float)kD;
#pragma unroll
  for (int r = 0; r < 8; ++r) {
    int i = ibase + r;
    atomicAdd(&lagAcc[i - j + 15], acc[r] * invD);
  }
  __syncthreads();

  if (tid < 31) {
    int lag = 16 * dt + (tid - 15);
    if (lag < 0) lag += kL;
    if (lag >= kL) lag -= kL;
    atomicAdd(&mean_corr[b * kL + lag], lagAcc[tid]);
  }
}

// ---------------------------------------------------------------------------
// Kernel 2: global mean over batches, iterative top-7 argmax, per-batch
// softmax of mean_corr[b, topk]. Single workgroup.
// ---------------------------------------------------------------------------
__global__ __launch_bounds__(256) void ac_topk_softmax_kernel(
    const float* __restrict__ mean_corr, int* __restrict__ topk,
    float* __restrict__ weights) {
  __shared__ float gm[kL];
  __shared__ float bval[256];
  __shared__ int   bidx[256];
  __shared__ int   chosen[kTOPK];
  const int tid = threadIdx.x;

  for (int l = tid; l < kL; l += 256) {
    float s = 0.0f;
    for (int b = 0; b < kB; ++b) s += mean_corr[b * kL + l];
    gm[l] = s * (1.0f / (float)kB);
  }
  __syncthreads();

  for (int kk = 0; kk < kTOPK; ++kk) {
    float best = -3.4e38f;
    int   bi   = 0;
    for (int l = tid; l < kL; l += 256) {
      if (gm[l] > best) { best = gm[l]; bi = l; }
    }
    bval[tid] = best;
    bidx[tid] = bi;
    __syncthreads();
    if (tid == 0) {
      float bb = bval[0];
      int   ii = bidx[0];
      for (int t = 1; t < 256; ++t)
        if (bval[t] > bb) { bb = bval[t]; ii = bidx[t]; }
      chosen[kk] = ii;
      topk[kk]   = ii;
      gm[ii]     = -3.4e38f;
    }
    __syncthreads();
  }

  if (tid < kB) {
    float w[kTOPK];
    float mx = -3.4e38f;
#pragma unroll
    for (int kk = 0; kk < kTOPK; ++kk) {
      w[kk] = mean_corr[tid * kL + chosen[kk]];
      mx    = fmaxf(mx, w[kk]);
    }
    float s = 0.0f;
#pragma unroll
    for (int kk = 0; kk < kTOPK; ++kk) {
      w[kk] = __expf(w[kk] - mx);
      s += w[kk];
    }
    float inv = 1.0f / s;
#pragma unroll
    for (int kk = 0; kk < kTOPK; ++kk)
      weights[tid * kTOPK + kk] = w[kk] * inv;
  }
}

// ---------------------------------------------------------------------------
// Kernel 3: out[b,l,:] = sum_k w[b,k] * V[b,(l - delay_k) mod L, :]
// TDM version, double-buffered: one workgroup per (8-row l-tile, b). Per tap
// the source is a contiguous 8x512 f32 tile (two descriptors if the circular
// shift wraps). Tap kk+1 is issued into the ping-pong buffer before waiting
// (s_wait_tensorcnt <= #descriptors of tap kk+1; TENSORcnt completes in-order
// per wave), so the tensor DMA overlaps the FMA/accumulate phase.
// ---------------------------------------------------------------------------
#if AC_HAS_TDM

__device__ __forceinline__ void ac_tdm_load_rows(const float* gsrc,
                                                 float* lds_dst, int nrows) {
  unsigned long long ga = (unsigned long long)(uintptr_t)gsrc;
  unsigned int       la = (unsigned int)(uintptr_t)lds_dst;
  // D# group0: count=1 | lds_addr | global_addr[56:0] | type=2
  v4u g0 = {1u, la, (unsigned int)ga,
            (unsigned int)(ga >> 32) | (2u << 30)};
  // D# group1: data_size=2 (4B); tensor_dim0=512; tensor_dim1=0x100000;
  //            tile_dim0=512; tile_dim1=nrows; tensor_dim0_stride=512
  v8i g1 = {(int)(2u << 16),                       // workgroup_mask=0, 4B elems
            (int)(512u << 16),                     // tensor_dim0 lo16 in [31:16]
            0,                                     // tensor_dim0 hi / dim1 lo
            (int)(0x10u | (512u << 16)),           // tensor_dim1 hi | tile_dim0
            nrows,                                 // tile_dim1 (tile_dim2=0)
            512,                                   // tensor_dim0_stride lo32
            0, 0};
  v4i g2 = {0, 0, 0, 0};
  v4i g3 = {0, 0, 0, 0};
  v8i g4 = {0, 0, 0, 0, 0, 0, 0, 0};  // 6-arg form (clang-23 / therock-10.0)
  __builtin_amdgcn_tensor_load_to_lds(g0, g1, g2, g3, g4, 0);
}

__global__ __launch_bounds__(256) void ac_gather_tdm_kernel(
    const float* __restrict__ v, const int* __restrict__ topk,
    const float* __restrict__ weights, float* __restrict__ out) {
  __shared__ float stage[2][kGR * kD];  // 2 x 16 KB ping-pong staging tiles
  __shared__ int   sDelay[kTOPK];
  __shared__ float sW[kTOPK];
  const int tid = threadIdx.x;
  const int b   = blockIdx.y;
  const int t0  = blockIdx.x * kGR;     // first output row of this tile

  if (tid < kTOPK) {
    sDelay[tid] = topk[tid];
    sW[tid]     = weights[b * kTOPK + tid];
  }
  __syncthreads();

  const float* vb = v + (size_t)b * kL * kD;

  // Issue one tap's (possibly wrap-split) tile into a staging buffer.
  // Returns the number of TDM descriptors issued (1 or 2).
  auto issue_tap = [&](int kk, float* buf) -> int {
    int s0 = t0 - sDelay[kk];
    if (s0 < 0) s0 += kL;
    int n = kL - s0;
    if (n > kGR) n = kGR;
    ac_tdm_load_rows(vb + (size_t)s0 * kD, buf, n);
    if (n < kGR) {
      ac_tdm_load_rows(vb, buf + n * kD, kGR - n);
      return 2;
    }
    return 1;
  };

  if (tid < 32) (void)issue_tap(0, stage[0]);  // wave 0 drives the TDM

  float4 acc[4] = {};

  for (int kk = 0; kk < kTOPK; ++kk) {
    if (tid < 32) {
      int nxt = 0;
      if (kk + 1 < kTOPK) nxt = issue_tap(kk + 1, stage[(kk + 1) & 1]);
      // wait for tap kk only (leave tap kk+1's descriptors in flight)
      if (nxt == 0)
        __builtin_amdgcn_s_wait_tensorcnt(0);
      else if (nxt == 1)
        __builtin_amdgcn_s_wait_tensorcnt(1);
      else
        __builtin_amdgcn_s_wait_tensorcnt(2);
    }
    __syncthreads();  // tap kk resident + visible to all waves
    const float4* st4 = (const float4*)stage[kk & 1];
    float         w   = sW[kk];
#pragma unroll
    for (int r = 0; r < 4; ++r) {
      float4 val = st4[tid + 256 * r];
      acc[r].x += w * val.x;
      acc[r].y += w * val.y;
      acc[r].z += w * val.z;
      acc[r].w += w * val.w;
    }
    __syncthreads();  // reads done before tap kk+2 overwrites this buffer
  }

  float4* out4 = (float4*)(out + ((size_t)b * kL + t0) * kD);
#pragma unroll
  for (int r = 0; r < 4; ++r) out4[tid + 256 * r] = acc[r];
}

#else  // !AC_HAS_TDM fallback: per-lane vectorized gather

__global__ __launch_bounds__(256) void ac_gather_combine_kernel(
    const float4* __restrict__ v4, const int* __restrict__ topk,
    const float* __restrict__ weights, float4* __restrict__ out4) {
  __shared__ int   sDelay[kTOPK];
  __shared__ float sW[kB * kTOPK];
  const int tid = threadIdx.x;
  if (tid < kTOPK) sDelay[tid] = topk[tid];
  if (tid < kB * kTOPK) sW[tid] = weights[tid];
  __syncthreads();

  constexpr int D4 = kD / 4;  // 128
  long g    = (long)blockIdx.x * 256 + tid;
  int  d4   = (int)(g & (D4 - 1));
  long rest = g >> 7;
  int  l    = (int)(rest % kL);
  int  b    = (int)(rest / kL);

  float4 sum = {0.0f, 0.0f, 0.0f, 0.0f};
#pragma unroll
  for (int kk = 0; kk < kTOPK; ++kk) {
    int sl = l - sDelay[kk];
    if (sl < 0) sl += kL;
    float  w   = sW[b * kTOPK + kk];
    float4 val = v4[(long)(b * kL + sl) * D4 + d4];
    sum.x += w * val.x;
    sum.y += w * val.y;
    sum.z += w * val.z;
    sum.w += w * val.w;
  }
  out4[g] = sum;
}

#endif  // AC_HAS_TDM

// ---------------------------------------------------------------------------
extern "C" void kernel_launch(void* const* d_in, const int* in_sizes, int n_in,
                              void* d_out, int out_size, void* d_ws,
                              size_t ws_size, hipStream_t stream) {
  (void)in_sizes; (void)n_in; (void)out_size; (void)ws_size;
  const float* q = (const float*)d_in[0];
  const float* k = (const float*)d_in[1];
  const float* v = (const float*)d_in[2];

  float* ws        = (float*)d_ws;
  float* mean_corr = ws;                      // kB*kL = 24576 floats
  int*   topk      = (int*)(ws + kB * kL);    // 7 ints
  float* weights   = ws + kB * kL + 16;       // kB*kTOPK floats

  ac_zero_kernel<<<(kB * kL + 255) / 256, 256, 0, stream>>>(mean_corr,
                                                            kB * kL);

  dim3 gridA(kNT, kB);
  ac_corr_kernel<<<gridA, 256, 0, stream>>>(q, k, mean_corr);

  ac_topk_softmax_kernel<<<1, 256, 0, stream>>>(mean_corr, topk, weights);

#if AC_HAS_TDM
  dim3 gridG(kL / kGR, kB);  // 192 x 16
  ac_gather_tdm_kernel<<<gridG, 256, 0, stream>>>(v, topk, weights,
                                                  (float*)d_out);
#else
  long total4 = (long)kB * kL * (kD / 4);  // 3,145,728 float4
  ac_gather_combine_kernel<<<(int)(total4 / 256), 256, 0, stream>>>(
      (const float4*)v, topk, weights, (float4*)d_out);
#endif
}
